// ValueGRUCUDA_5909874999720
// MI455X (gfx1250) — compile-verified, matrix operand order checked by reading, not run on hardware
//
#include <hip/hip_runtime.h>
#include <hip/hip_bf16.h>

// ---------------------------------------------------------------------------
// ValueGRU for MI455X (gfx1250): bf16 WMMA + async-to-LDS weight staging.
//   K1: convert weights + initial hidden to bf16 workspace
//   K2: X = relu(obs @ W_embed^T + b_embed)  -> bf16  (WMMA, K=128)
//   K3 (x512, sequential): per block (8 waves), stage W_ih/W_hh column tiles
//       into LDS with GLOBAL_LOAD_ASYNC_TO_LDS_B128 (ASYNCcnt), then fused
//       gx_t + gh_t GEMMs: 48 WMMAs/wave fed by ds_load_b128 (conflict-free
//       528B row stride), gate math, double-buffered h (f32 + bf16 mirrors)
//   K4: value head for t=T-1 + emit hidden_new
// ---------------------------------------------------------------------------

typedef __bf16 bf16_t;
typedef __attribute__((ext_vector_type(16))) __bf16 v16bf;
typedef __attribute__((ext_vector_type(8)))  float  v8f;
typedef __attribute__((ext_vector_type(4)))  unsigned int u32x4;
typedef __attribute__((ext_vector_type(4)))  float  f32x4;

// GCC-style int4 vector matching the async-builtin's pointee type
typedef int v4i __attribute__((vector_size(16)));
typedef __attribute__((address_space(1))) v4i as1_v4i;
typedef __attribute__((address_space(3))) v4i as3_v4i;

#define OBSD 128
#define HID  256
#define NB   256
#define TT   512

// padded LDS row stride for the 256-element bf16 weight rows:
// 528 B = 132 dwords; 132 mod 64 = 4 -> 16 lanes hit 16 distinct bank groups.
#define LDS_ROW_B 528
#define LDS_MAT_B (48 * LDS_ROW_B)   // 3 gates x 16 rows

// ---- CDNA5 async copy: global -> LDS (ASYNCcnt) ----------------------------

__device__ __forceinline__ void async_cp16(void* lds_dst, const void* gsrc) {
#if defined(__gfx1250__) && __has_builtin(__builtin_amdgcn_global_load_async_to_lds_b128)
  __builtin_amdgcn_global_load_async_to_lds_b128(
      (as1_v4i*)(unsigned long long)(size_t)gsrc,
      (as3_v4i*)(unsigned int)(size_t)lds_dst,
      0, 0);
#else
  asm volatile("global_load_async_to_lds_b128 %0, %1, off"
               :: "v"((unsigned int)(size_t)lds_dst),
                  "v"((unsigned long long)(size_t)gsrc)
               : "memory");
#endif
}

__device__ __forceinline__ void async_wait0() {
#if defined(__gfx1250__) && __has_builtin(__builtin_amdgcn_s_wait_asynccnt)
  __builtin_amdgcn_s_wait_asynccnt(0);
#else
  asm volatile("s_wait_asynccnt 0x0" ::: "memory");
#endif
}

// ---- helpers --------------------------------------------------------------

__device__ __forceinline__ v16bf load_tile16(const bf16_t* p) {
  // 16B-aligned; elements [0..7] = p[0..7], [8..15] = p[16..23]
  union { u32x4 u[2]; v16bf v; } t;
  t.u[0] = *(const u32x4*)(p);
  t.u[1] = *(const u32x4*)(p + 16);
  return t.v;
}

__device__ __forceinline__ v16bf load_tile16_lds(const char* p) {
  union { u32x4 u[2]; v16bf v; } t;
  t.u[0] = *(const u32x4*)(p);
  t.u[1] = *(const u32x4*)(p + 32);   // +16 bf16 elements
  return t.v;
}

__device__ __forceinline__ float sigmoid_f(float x) {
  return 1.0f / (1.0f + __expf(-x));
}

__device__ __forceinline__ float tanh_f(float x) {
  float a = fabsf(x);
  float t = 1.0f - 2.0f / (__expf(2.0f * a) + 1.0f);
  return copysignf(t, x);
}

__device__ __forceinline__ v8f wmma_bf16(v16bf a, v16bf b, v8f c) {
  return __builtin_amdgcn_wmma_f32_16x16x32_bf16(
      /*neg_a=*/false, a, /*neg_b=*/false, b,
      /*c_mod=*/(short)0, c, /*reuse_a=*/false, /*reuse_b=*/false);
}

// ---- K1: convert weights / init hidden ------------------------------------

__global__ void k_convert(const float* __restrict__ We,
                          const float* __restrict__ Wih,
                          const float* __restrict__ Whh,
                          const float* __restrict__ hid0,
                          bf16_t* __restrict__ We_b,
                          bf16_t* __restrict__ Wih_b,
                          bf16_t* __restrict__ Whh_b,
                          float*  __restrict__ h32,
                          bf16_t* __restrict__ hbf) {
  int i = blockIdx.x * blockDim.x + threadIdx.x;
  int stride = gridDim.x * blockDim.x;
  for (int j = i; j < HID * OBSD; j += stride) We_b[j] = (bf16_t)We[j];
  for (int j = i; j < 3 * HID * HID; j += stride) {
    Wih_b[j] = (bf16_t)Wih[j];
    Whh_b[j] = (bf16_t)Whh[j];
  }
  for (int j = i; j < NB * HID; j += stride) {
    float v = hid0[j];
    h32[j] = v;
    hbf[j] = (bf16_t)v;
  }
}

// ---- K2: embedding GEMM, relu, bf16 out -----------------------------------
// grid = (NB*TT/16, HID/64), block = 32 (one wave). 16 rows x 64 cols/wave.

__global__ void k_embed(const float* __restrict__ obs,
                        const bf16_t* __restrict__ Web,
                        const float* __restrict__ b_embed,
                        bf16_t* __restrict__ Xb) {
  const int lane = threadIdx.x;
  const int r  = lane & 15;
  const int hi = lane >> 4;
  const int m0 = blockIdx.x * 16;
  const int g0 = blockIdx.y * 64;

  v8f acc[4];
#pragma unroll
  for (int j = 0; j < 4; j++) acc[j] = (v8f){0.f,0.f,0.f,0.f,0.f,0.f,0.f,0.f};

  const float* arow = obs + (size_t)(m0 + r) * OBSD;
#pragma unroll
  for (int kk = 0; kk < OBSD; kk += 32) {
    const int kb = kk + hi * 8;
    f32x4 x0 = *(const f32x4*)(arow + kb);
    f32x4 x1 = *(const f32x4*)(arow + kb + 4);
    f32x4 x2 = *(const f32x4*)(arow + kb + 16);
    f32x4 x3 = *(const f32x4*)(arow + kb + 20);
    v16bf a;
#pragma unroll
    for (int e = 0; e < 4; e++) {
      a[e]      = (bf16_t)x0[e];
      a[4 + e]  = (bf16_t)x1[e];
      a[8 + e]  = (bf16_t)x2[e];
      a[12 + e] = (bf16_t)x3[e];
    }
#pragma unroll
    for (int j = 0; j < 4; j++) {
      const bf16_t* brow = Web + (size_t)(g0 + j * 16 + r) * OBSD + kb;
      v16bf b = load_tile16(brow);
      acc[j] = wmma_bf16(a, b, acc[j]);
    }
  }

#pragma unroll
  for (int j = 0; j < 4; j++) {
    const int g = g0 + j * 16 + r;
    const float bias = b_embed[g];
#pragma unroll
    for (int v = 0; v < 8; v++) {
      const int m = m0 + v + 8 * hi;
      float val = acc[j][v] + bias;
      val = val > 0.f ? val : 0.f;
      Xb[(size_t)m * HID + g] = (bf16_t)val;
    }
  }
}

// ---- K3: one GRU timestep --------------------------------------------------
// grid = (NB/128, HID/16), block = 256 (8 waves).
// Block stages W_ih/W_hh tiles for its 16 output columns (all 3 gates, full
// K=256) into LDS via async b128 copies; 8 waves each own 16 batch rows.

__global__ void __launch_bounds__(256)
k_step(const bf16_t* __restrict__ Xb,
       const bf16_t* __restrict__ Wihb,
       const bf16_t* __restrict__ Whhb,
       const float*  __restrict__ b_ih,
       const float*  __restrict__ b_hh,
       const float*  __restrict__ h32_prev,
       const bf16_t* __restrict__ hbf_prev,
       float*  __restrict__ h32_next,
       bf16_t* __restrict__ hbf_next,
       const float* __restrict__ W_val,
       const float* __restrict__ b_val,
       float* __restrict__ values,
       int t) {
  __shared__ __align__(16) char lds_w[2 * LDS_MAT_B];   // ~50.7 KB

  const int tid  = threadIdx.x;
  const int lane = tid & 31;
  const int wave = tid >> 5;
  const int r  = lane & 15;
  const int hi = lane >> 4;
  const int n0 = blockIdx.x * 128 + wave * 16;
  const int g0 = blockIdx.y * 16;

  // ---- stage weights: 48 rows x 512B per matrix, 16B chunks, async --------
  for (int c = tid; c < 1536; c += 256) {          // 6 iterations
    const int m    = c >> 5;                        // row 0..47 (= q*16 + j)
    const int part = c & 31;                        // 16B chunk in row
    const int q = m >> 4, j = m & 15;
    const bf16_t* si = Wihb + (size_t)(q * HID + g0 + j) * HID + part * 8;
    const bf16_t* sh = Whhb + (size_t)(q * HID + g0 + j) * HID + part * 8;
    char* di = lds_w + m * LDS_ROW_B + part * 16;
    char* dh = lds_w + LDS_MAT_B + m * LDS_ROW_B + part * 16;
    async_cp16(di, si);
    async_cp16(dh, sh);
  }
  async_wait0();
  __syncthreads();

  // ---- accumulators seeded with biases (gx: b_ih, gh: b_hh, kept split) ---
  v8f ax[3], ah[3];
#pragma unroll
  for (int q = 0; q < 3; q++) {
    const float bi = b_ih[q * HID + g0 + r];
    const float bh = b_hh[q * HID + g0 + r];
    ax[q] = (v8f){bi,bi,bi,bi,bi,bi,bi,bi};
    ah[q] = (v8f){bh,bh,bh,bh,bh,bh,bh,bh};
  }

  const bf16_t* xrow = Xb + ((size_t)(n0 + r) * TT + t) * HID;
  const bf16_t* hrow = hbf_prev + (size_t)(n0 + r) * HID;

#pragma unroll
  for (int kk = 0; kk < HID; kk += 32) {
    const int kb = kk + hi * 8;
    v16bf a_x = load_tile16(xrow + kb);
    v16bf a_h = load_tile16(hrow + kb);
#pragma unroll
    for (int q = 0; q < 3; q++) {
      const char* wi = lds_w + (q * 16 + r) * LDS_ROW_B + kb * 2;
      const char* wh = wi + LDS_MAT_B;
      v16bf b_i = load_tile16_lds(wi);
      v16bf b_h = load_tile16_lds(wh);
      ax[q] = wmma_bf16(a_x, b_i, ax[q]);
      ah[q] = wmma_bf16(a_h, b_h, ah[q]);
    }
  }

  const int g = g0 + r;
#pragma unroll
  for (int v = 0; v < 8; v++) {
    const int n = n0 + v + 8 * hi;
    const float hp = h32_prev[(size_t)n * HID + g];
    const float rg = sigmoid_f(ax[0][v] + ah[0][v]);
    const float zg = sigmoid_f(ax[1][v] + ah[1][v]);
    const float ng = tanh_f(ax[2][v] + rg * ah[2][v]);
    const float hn = (1.0f - zg) * ng + zg * hp;
    h32_next[(size_t)n * HID + g] = hn;
    hbf_next[(size_t)n * HID + g] = (bf16_t)hn;
  }

  // value head for timestep t-1 (h_prev == hs[t-1]); only col-tile 0 blocks.
  if (blockIdx.y == 0 && t > 0) {
    const int n = n0 + r;
    const float* hv = h32_prev + (size_t)n * HID + hi * 128;
    const float* wv = W_val + hi * 128;
    float s = 0.f;
#pragma unroll 8
    for (int k = 0; k < 128; k++) s += hv[k] * wv[k];
    s += __shfl_xor(s, 16, 32);
    if (hi == 0) values[(size_t)n * TT + (t - 1)] = s + b_val[0];
  }
}

// ---- K4: last value + hidden_new output -----------------------------------

__global__ void k_final(const float* __restrict__ h_last,
                        const float* __restrict__ W_val,
                        const float* __restrict__ b_val,
                        float* __restrict__ out) {
  const int gid = blockIdx.x * blockDim.x + threadIdx.x;
  if (gid < NB * HID) out[NB * TT + gid] = h_last[gid];
  if (gid < NB) {
    const float* hv = h_last + (size_t)gid * HID;
    float s = b_val[0];
#pragma unroll 8
    for (int k = 0; k < HID; k++) s += hv[k] * W_val[k];
    out[(size_t)gid * TT + (TT - 1)] = s;
  }
}

// ---- launch ----------------------------------------------------------------

extern "C" void kernel_launch(void* const* d_in, const int* in_sizes, int n_in,
                              void* d_out, int out_size, void* d_ws, size_t ws_size,
                              hipStream_t stream) {
  const float* obs    = (const float*)d_in[0];
  const float* hidden = (const float*)d_in[1];
  const float* W_emb  = (const float*)d_in[2];
  const float* b_emb  = (const float*)d_in[3];
  const float* W_ih   = (const float*)d_in[4];
  const float* W_hh   = (const float*)d_in[5];
  const float* b_ih   = (const float*)d_in[6];
  const float* b_hh   = (const float*)d_in[7];
  const float* W_val  = (const float*)d_in[8];
  const float* b_val  = (const float*)d_in[9];
  float* out = (float*)d_out;

  // workspace carve-up (256B aligned chunks), ~69 MB total
  size_t off = 0;
  auto carve = [&](size_t bytes) -> void* {
    void* p = (char*)d_ws + off;
    off += (bytes + 255) & ~(size_t)255;
    return p;
  };
  bf16_t* We_b  = (bf16_t*)carve((size_t)HID * OBSD * 2);
  bf16_t* Wih_b = (bf16_t*)carve((size_t)3 * HID * HID * 2);
  bf16_t* Whh_b = (bf16_t*)carve((size_t)3 * HID * HID * 2);
  bf16_t* Xb    = (bf16_t*)carve((size_t)NB * TT * HID * 2);
  float*  h32a  = (float*) carve((size_t)NB * HID * 4);
  float*  h32b  = (float*) carve((size_t)NB * HID * 4);
  bf16_t* hbfa  = (bf16_t*)carve((size_t)NB * HID * 2);
  bf16_t* hbfb  = (bf16_t*)carve((size_t)NB * HID * 2);

  k_convert<<<512, 256, 0, stream>>>(W_emb, W_ih, W_hh, hidden,
                                     We_b, Wih_b, Whh_b, h32a, hbfa);

  k_embed<<<dim3(NB * TT / 16, HID / 64), 32, 0, stream>>>(obs, We_b, b_emb, Xb);

  for (int t = 0; t < TT; t++) {
    const bool even = (t & 1) == 0;
    const float*  hp  = even ? h32a : h32b;
    const bf16_t* hpb = even ? hbfa : hbfb;
    float*  hn  = even ? h32b : h32a;
    bf16_t* hnb = even ? hbfb : hbfa;
    k_step<<<dim3(NB / 128, HID / 16), 256, 0, stream>>>(
        Xb, Wih_b, Whh_b, b_ih, b_hh, hp, hpb, hn, hnb, W_val, b_val, out, t);
  }

  // TT even: final h landed in buffer a
  k_final<<<256, 256, 0, stream>>>(h32a, W_val, b_val, out);
}
